// Encoder_87454124082267
// MI455X (gfx1250) — compile-verified, matrix operand order checked by reading, not run on hardware
//
#include <hip/hip_runtime.h>
#include <hip/hip_bf16.h>

typedef __attribute__((ext_vector_type(16))) __bf16 v16bf;
typedef __attribute__((ext_vector_type(8)))  float  v8f;

#define VOCAB   32000
#define EMB     256
#define UNITS   1024
#define BATCH   128
#define SEQ     256
#define N3      (3*UNITS)     // 3072
#define NT3     (N3/16)       // 192 column tiles across all 3 gates
#define NTH     (UNITS/16)    // 64 column tiles per gate
#define KKW     (EMB/32)      // 8  K-chunks for the W GEMM
#define KKU     (UNITS/32)    // 32 K-chunks for the U GEMM
#define MT      (BATCH/16)    // 8  row tiles
#define LANE_ELEMS 512        // 32 lanes * 16 bf16 per 16x16x32 operand chunk

// K offset inside a 32-wide chunk for (lane-half, element e) per the CDNA5
// 16-bit A-matrix VGPR layout (pairs contiguous in K).
__device__ __forceinline__ int koff(int half, int e) {
    int vg = e >> 1, pos = e & 1;
    return (vg < 4) ? (half * 8 + vg * 2 + pos)
                    : (16 + half * 8 + (vg - 4) * 2 + pos);
}

// Pack a KxN3 f32 weight matrix into WMMA B-operand order:
// dst[kkIdx][ntile][lane][16] (each lane's 16 bf16 contiguous = 32B load).
__global__ void pack_weights_kernel(const float* __restrict__ src,
                                    __bf16* __restrict__ dst, int kkTiles) {
    size_t total = (size_t)kkTiles * NT3 * LANE_ELEMS;
    for (size_t idx = (size_t)blockIdx.x * blockDim.x + threadIdx.x;
         idx < total; idx += (size_t)gridDim.x * blockDim.x) {
        int e     = (int)(idx & 15);
        int lane  = (int)((idx >> 4) & 31);
        size_t t2 = idx >> 9;
        int ntile = (int)(t2 % NT3);
        int kkIdx = (int)(t2 / NT3);
        int K   = kkIdx * 32 + koff(lane >> 4, e);
        int col = ntile * 16 + (lane & 15);
        dst[idx] = (__bf16)src[(size_t)K * N3 + col];
    }
}

// Gather embedding rows per token and pack into WMMA A-operand order:
// eA[t][kkIdx][mTile][lane][16].
__global__ void embed_pack_kernel(const int* __restrict__ x,
                                  const float* __restrict__ emb,
                                  __bf16* __restrict__ eA) {
    size_t total = (size_t)SEQ * KKW * MT * LANE_ELEMS;
    for (size_t idx = (size_t)blockIdx.x * blockDim.x + threadIdx.x;
         idx < total; idx += (size_t)gridDim.x * blockDim.x) {
        int e     = (int)(idx & 15);
        int lane  = (int)((idx >> 4) & 31);
        size_t t2 = idx >> 9;
        int mTile = (int)(t2 % MT); t2 /= MT;
        int kkIdx = (int)(t2 % KKW);
        int t     = (int)(t2 / KKW);
        int brow  = mTile * 16 + (lane & 15);
        int K     = kkIdx * 32 + koff(lane >> 4, e);
        int tok   = x[brow * SEQ + t];          // x is (B,S) row-major
        eA[idx] = (__bf16)emb[(size_t)tok * EMB + K];
    }
}

// Pack the initial f32 hidden state (BATCH x UNITS) into WMMA A-operand order:
// hA[kkIdx][mTile][lane][16].
__global__ void pack_hidden_kernel(const float* __restrict__ h,
                                   __bf16* __restrict__ hA) {
    size_t total = (size_t)KKU * MT * LANE_ELEMS;
    for (size_t idx = (size_t)blockIdx.x * blockDim.x + threadIdx.x;
         idx < total; idx += (size_t)gridDim.x * blockDim.x) {
        int e     = (int)(idx & 15);
        int lane  = (int)((idx >> 4) & 31);
        size_t t2 = idx >> 9;
        int mTile = (int)(t2 % MT);
        int kkIdx = (int)(t2 / MT);
        int row = mTile * 16 + (lane & 15);
        int K   = kkIdx * 32 + koff(lane >> 4, e);
        hA[idx] = (__bf16)h[(size_t)row * UNITS + K];
    }
}

// One GRU time step. Each wave computes one 16x16 tile of h_new:
// 4 accumulators (z, r, xh, rh) via fused e@W (K=256) + h@U (K=1024) bf16 WMMA.
// A operands for both GEMMs come pre-packed (eA, hAprev); this wave also
// scatters h_new as bf16 into hAnext so the next step needs no conversion.
__global__ __launch_bounds__(128)
void gru_step_kernel(const __bf16* __restrict__ eA,
                     const __bf16* __restrict__ Wp,
                     const __bf16* __restrict__ Up,
                     const float*  __restrict__ bvec,
                     const float*  __restrict__ hprev,   // f32 state (read)
                     float*        __restrict__ hnext,   // f32 state (write)
                     const __bf16* __restrict__ hAprev,  // packed A (read)
                     __bf16*       __restrict__ hAnext,  // packed A (write)
                     float*        __restrict__ out,
                     int t, int write_state) {
    const int lane  = threadIdx.x & 31;
    const int w     = blockIdx.x * (blockDim.x >> 5) + (threadIdx.x >> 5);
    const int mTile = w >> 6;      // 0..7
    const int nt    = w & 63;      // 0..63 (column tile within one gate)
    const int half  = lane >> 4;
    const int col   = nt * 16 + (lane & 15);   // column in [0, UNITS)

    const float* b0 = bvec;
    const float* b1 = bvec + N3;
    const float bz  = b0[col]             + b1[col];
    const float br  = b0[UNITS + col]     + b1[UNITS + col];
    const float bxh = b0[2 * UNITS + col];
    const float brh = b1[2 * UNITS + col];

    v8f accz, accr, accxh, accrh;
#pragma unroll
    for (int i = 0; i < 8; ++i) { accz[i] = bz; accr[i] = br; accxh[i] = bxh; accrh[i] = brh; }

    // ---- input projection part: A = packed embeddings, K = EMB = 256 ----
#pragma unroll
    for (int kkIdx = 0; kkIdx < KKW; ++kkIdx) {
        v16bf a  = *(const v16bf*)(eA + ((((size_t)t * KKW + kkIdx) * MT + mTile) << 9) + lane * 16);
        v16bf wz = *(const v16bf*)(Wp + (((size_t)kkIdx * NT3 + nt)           << 9) + lane * 16);
        v16bf wr = *(const v16bf*)(Wp + (((size_t)kkIdx * NT3 + NTH + nt)     << 9) + lane * 16);
        v16bf wh = *(const v16bf*)(Wp + (((size_t)kkIdx * NT3 + 2 * NTH + nt) << 9) + lane * 16);
        accz  = __builtin_amdgcn_wmma_f32_16x16x32_bf16(false, a, false, wz, (short)0, accz,  false, false);
        accr  = __builtin_amdgcn_wmma_f32_16x16x32_bf16(false, a, false, wr, (short)0, accr,  false, false);
        accxh = __builtin_amdgcn_wmma_f32_16x16x32_bf16(false, a, false, wh, (short)0, accxh, false, false);
    }

    // ---- recurrent part: A = pre-packed bf16 h_prev, K = UNITS = 1024 ----
#pragma unroll 4
    for (int kkIdx = 0; kkIdx < KKU; ++kkIdx) {
        v16bf a  = *(const v16bf*)(hAprev + (((size_t)kkIdx * MT + mTile)     << 9) + lane * 16);
        v16bf uz = *(const v16bf*)(Up + (((size_t)kkIdx * NT3 + nt)           << 9) + lane * 16);
        v16bf ur = *(const v16bf*)(Up + (((size_t)kkIdx * NT3 + NTH + nt)     << 9) + lane * 16);
        v16bf uh = *(const v16bf*)(Up + (((size_t)kkIdx * NT3 + 2 * NTH + nt) << 9) + lane * 16);
        accz  = __builtin_amdgcn_wmma_f32_16x16x32_bf16(false, a, false, uz, (short)0, accz,  false, false);
        accr  = __builtin_amdgcn_wmma_f32_16x16x32_bf16(false, a, false, ur, (short)0, accr,  false, false);
        accrh = __builtin_amdgcn_wmma_f32_16x16x32_bf16(false, a, false, uh, (short)0, accrh, false, false);
    }

    // ---- gate math + stores (C/D layout: lane col = L%16, row = vr + 8*half) ----
    // Scatter target for h_new in packed A layout (col-derived parts invariant):
    const int c32   = col & 31;
    const int kkO   = col >> 5;                       // K-chunk of this column
    const int halfA = (c32 >> 3) & 1;                 // which lane-half holds it
    const int posA  = c32 & 1;
    const int vgA   = (c32 < 16) ? ((c32 & 7) >> 1) : (4 + (((c32 - 16) & 7) >> 1));
    const int eA_i  = 2 * vgA + posA;
    __bf16* hAdst = hAnext + (((size_t)kkO * MT + mTile) << 9) + (halfA * 16) * 16 + eA_i;

#pragma unroll
    for (int vr = 0; vr < 8; ++vr) {
        int rlo   = half * 8 + vr;                 // row within the 16-row tile
        int row   = mTile * 16 + rlo;              // batch index
        float z   = 1.0f / (1.0f + __expf(-accz[vr]));
        float r   = 1.0f / (1.0f + __expf(-accr[vr]));
        float hh  = tanhf(accxh[vr] + r * accrh[vr]);
        float hol = hprev[(size_t)row * UNITS + col];
        float hn  = z * hol + (1.0f - z) * hh;
        hnext[(size_t)row * UNITS + col] = hn;
        hAdst[rlo * 16] = (__bf16)hn;              // lane' = halfA*16 + rlo
        out[(size_t)row * SEQ * UNITS + (size_t)t * UNITS + col] = hn;
        if (write_state)
            out[(size_t)BATCH * SEQ * UNITS + (size_t)row * UNITS + col] = hn;
    }
}

extern "C" void kernel_launch(void* const* d_in, const int* in_sizes, int n_in,
                              void* d_out, int out_size, void* d_ws, size_t ws_size,
                              hipStream_t stream) {
    const int*   x      = (const int*)  d_in[0];
    const float* hidden = (const float*)d_in[1];
    const float* emb    = (const float*)d_in[2];
    const float* W      = (const float*)d_in[3];
    const float* U      = (const float*)d_in[4];
    const float* bvec   = (const float*)d_in[5];
    float* out = (float*)d_out;

    // Workspace carve-up (~25 MB total, 256B aligned slices).
    char* ws = (char*)d_ws;
    auto carve = [&](size_t bytes) {
        char* p = ws;
        ws += (bytes + 255) & ~(size_t)255;
        return p;
    };
    __bf16* Wp  = (__bf16*)carve((size_t)KKW * NT3 * LANE_ELEMS * sizeof(__bf16)); // 1.5 MB
    __bf16* Up  = (__bf16*)carve((size_t)KKU * NT3 * LANE_ELEMS * sizeof(__bf16)); // 6 MB
    __bf16* eAb = (__bf16*)carve((size_t)SEQ * KKW * MT * LANE_ELEMS * sizeof(__bf16)); // 16 MB
    float*  h0  = (float*)carve((size_t)BATCH * UNITS * sizeof(float));            // 512 KB
    float*  h1  = (float*)carve((size_t)BATCH * UNITS * sizeof(float));            // 512 KB
    __bf16* hA0 = (__bf16*)carve((size_t)KKU * MT * LANE_ELEMS * sizeof(__bf16));  // 256 KB
    __bf16* hA1 = (__bf16*)carve((size_t)KKU * MT * LANE_ELEMS * sizeof(__bf16));  // 256 KB

    // One-time packs (parallel, fully independent).
    pack_weights_kernel<<<1024, 256, 0, stream>>>(W, Wp, KKW);
    pack_weights_kernel<<<4096, 256, 0, stream>>>(U, Up, KKU);
    embed_pack_kernel<<<8192, 256, 0, stream>>>(x, emb, eAb);
    pack_hidden_kernel<<<512, 256, 0, stream>>>(hidden, hA0);  // parity-0 buffer

    // Sequential recurrence: 256 graph-captured step launches, double-buffered
    // f32 state (gate math) + bf16 packed A state (WMMA operand).
    float*  hb[2]  = { h0, h1 };
    __bf16* hAb[2] = { hA0, hA1 };
    for (int t = 0; t < SEQ; ++t) {
        const float*  hp  = (t == 0) ? hidden : hb[t & 1];
        float*        hn  = hb[(t + 1) & 1];
        const __bf16* hAp = hAb[t & 1];
        __bf16*       hAn = hAb[(t + 1) & 1];
        gru_step_kernel<<<128, 128, 0, stream>>>(eAb, Wp, Up, bvec, hp, hn,
                                                 hAp, hAn, out, t,
                                                 (t == SEQ - 1) ? 1 : 0);
    }
}
